// Decoder_55387898249629
// MI455X (gfx1250) — matrix operand order repaired and checked by reading, not compiled, not run on hardware
//
#include <hip/hip_runtime.h>
#include <hip/hip_bf16.h>
#include <math.h>

#define NB   32          // batch
#define S0D  2132        // nurses+patients+clinics
#define STOT 2133        // + no_op token
#define SPAD 2144        // padded to /16 (134 tiles)
#define ED   512
#define QDIM 1024
#define KDH  128
#define NH   8
#define FFD  512
#define NL   6
#define NNUR 32
#define NEGV (-1000000000.0f)

typedef __attribute__((ext_vector_type(16))) __bf16 v16bf;
typedef __attribute__((ext_vector_type(8)))  __bf16 v8bf;
typedef __attribute__((ext_vector_type(8)))  float  v8f;

static __device__ __forceinline__ unsigned short f2bf(float f) {
    unsigned int u = __float_as_uint(f);
    u += 0x7FFFu + ((u >> 16) & 1u);           // round-to-nearest-even
    return (unsigned short)(u >> 16);
}
static __device__ __forceinline__ float bf2f(unsigned short u) {
    return __uint_as_float(((unsigned int)u) << 16);
}
static __device__ __forceinline__ v16bf cat8(v8bf lo, v8bf hi) {
    v16bf r;
#pragma unroll
    for (int i = 0; i < 8; ++i) { r[i] = lo[i]; r[i + 8] = hi[i]; }
    return r;
}

// ---------------- prep: h -> bf16 (padded), q0 ----------------
__global__ __launch_bounds__(256) void k_prep_h(const float* __restrict__ emb,
                                                const float* __restrict__ noop,
                                                unsigned short* __restrict__ hmat) {
    int blk = blockIdx.x;
    int b = blk / SPAD, s = blk % SPAD;
    unsigned short* orow = hmat + ((size_t)b * SPAD + s) * ED;
#pragma unroll
    for (int k = 0; k < 2; ++k) {
        int e = threadIdx.x + k * 256;
        float v;
        if (s < S0D)       v = emb[((size_t)b * S0D + s) * ED + e];
        else if (s == S0D) v = noop[e];
        else               v = 0.0f;
        orow[e] = f2bf(v);
    }
}

__global__ __launch_bounds__(256) void k_prep_q(const float* __restrict__ emb,
                                                const float* __restrict__ embm,
                                                const float* __restrict__ noop,
                                                const int* __restrict__ nidx,
                                                float* __restrict__ q) {
    int b = blockIdx.x, idx = nidx[0];
#pragma unroll
    for (int k = 0; k < 2; ++k) {
        int e = threadIdx.x + k * 256;
        q[(size_t)b * QDIM + e] = embm[(size_t)b * ED + e];
        float hv = (idx < S0D) ? emb[((size_t)b * S0D + idx) * ED + e] : noop[e];
        q[(size_t)b * QDIM + ED + e] = hv;
    }
}

// 16x16 tiled transpose: hT[b][e][s] = h[b][s][e]
__global__ __launch_bounds__(256) void k_transpose(const unsigned short* __restrict__ hmat,
                                                   unsigned short* __restrict__ hT) {
    __shared__ unsigned short tile[16][17];
    int b = blockIdx.x, s0 = blockIdx.y * 16, e0 = blockIdx.z * 16;
    int tx = threadIdx.x & 15, ty = threadIdx.x >> 4;
    tile[ty][tx] = hmat[((size_t)b * SPAD + s0 + ty) * ED + e0 + tx];
    __syncthreads();
    hT[((size_t)b * ED + e0 + ty) * SPAD + s0 + tx] = tile[tx][ty];
}

// ---------------- per-layer: fold Wq/Wk into one E-vector per (b,head) ----------------
__global__ __launch_bounds__(128) void k_qk(const float* __restrict__ q,
                                            const float* __restrict__ Wq_l,
                                            const float* __restrict__ Wk_l,
                                            unsigned short* __restrict__ qkt) {
    int b = blockIdx.x, h = blockIdx.y, tid = threadIdx.x;
    unsigned short* orow = qkt + ((size_t)b * 16 + h) * ED;
    if (h >= NH) {                                  // zero pad head rows (B-matrix cols 8..15)
        for (int e = tid; e < ED; e += 128) orow[e] = 0;
        return;
    }
    __shared__ float qs[QDIM];
    __shared__ float Qh[KDH];
    for (int d = tid; d < QDIM; d += 128) qs[d] = q[(size_t)b * QDIM + d];
    __syncthreads();
    {   // Qh[kd] = sum_d q[d] * Wq[h][d][kd]
        const float* w = Wq_l + (size_t)h * QDIM * KDH + tid;
        float acc = 0.f;
        for (int d = 0; d < QDIM; ++d) acc += qs[d] * w[(size_t)d * KDH];
        Qh[tid] = acc;
    }
    __syncthreads();
    const float scale = 0.08838834764831845f;       // 1/sqrt(KD), folded into scores
    for (int e = tid; e < ED; e += 128) {
        const float* w = Wk_l + ((size_t)h * ED + e) * KDH;
        float acc = 0.f;
#pragma unroll 8
        for (int kd = 0; kd < KDH; ++kd) acc += w[kd] * Qh[kd];
        orow[e] = f2bf(acc * scale);
    }
}

// ---------------- scores: (S x E) @ (E x 16heads) via WMMA bf16 ----------------
__global__ __launch_bounds__(32) void k_scores(const unsigned short* __restrict__ hmat,
                                               const unsigned short* __restrict__ qkt,
                                               const int* __restrict__ va,
                                               float* __restrict__ scores) {
    int b = blockIdx.x, s0 = blockIdx.y * 16;
    int lane = threadIdx.x, r16 = lane & 15, hi = lane >> 4;
    const unsigned short* arow = hmat + ((size_t)b * SPAD + s0 + r16) * ED + hi * 8;
    const unsigned short* brow = qkt + ((size_t)b * 16 + r16) * ED + hi * 8;
    v8f c = {0.f, 0.f, 0.f, 0.f, 0.f, 0.f, 0.f, 0.f};
#pragma unroll 4
    for (int e0 = 0; e0 < ED; e0 += 32) {
        v8bf alo = *(const v8bf*)(arow + e0);
        v8bf ahi = *(const v8bf*)(arow + e0 + 16);
        v8bf blo = *(const v8bf*)(brow + e0);
        v8bf bhi = *(const v8bf*)(brow + e0 + 16);
        c = __builtin_amdgcn_wmma_f32_16x16x32_bf16(false, cat8(alo, ahi),
                                                    false, cat8(blo, bhi),
                                                    (short)0, c, false, false);
    }
    int head = r16;
    if (head < NH) {
#pragma unroll
        for (int r = 0; r < 8; ++r) {
            int s = s0 + r + hi * 8;
            bool valid = (s >= NNUR) && (s < STOT) && (va[(size_t)b * (STOT - NNUR) + (s - NNUR)] != 0);
            scores[((size_t)b * NH + head) * SPAD + s] = valid ? c[r] : NEGV;
        }
    }
}

// ---------------- softmax over S per (b,head); writes bf16 attn (pad heads zeroed) ----------------
__global__ __launch_bounds__(256) void k_softmax(const float* __restrict__ scores,
                                                 unsigned short* __restrict__ attn) {
    int b = blockIdx.x, h = blockIdx.y, tid = threadIdx.x;
    unsigned short* orow = attn + ((size_t)b * 16 + h) * SPAD;
    if (h >= NH) {
        for (int s = tid; s < SPAD; s += 256) orow[s] = 0;
        return;
    }
    const float* srow = scores + ((size_t)b * NH + h) * SPAD;
    __shared__ float red[256];
    float mx = -3.0e38f;
    for (int s = tid; s < SPAD; s += 256) mx = fmaxf(mx, srow[s]);
    red[tid] = mx; __syncthreads();
    for (int o = 128; o > 0; o >>= 1) { if (tid < o) red[tid] = fmaxf(red[tid], red[tid + o]); __syncthreads(); }
    mx = red[0]; __syncthreads();
    float sm = 0.f;
    for (int s = tid; s < SPAD; s += 256) sm += __expf(srow[s] - mx);
    red[tid] = sm; __syncthreads();
    for (int o = 128; o > 0; o >>= 1) { if (tid < o) red[tid] += red[tid + o]; __syncthreads(); }
    float inv = 1.0f / red[0];
    for (int s = tid; s < SPAD; s += 256) orow[s] = f2bf(__expf(srow[s] - mx) * inv);
}

// ---------------- ctx: (16heads x S) @ (S x E) via WMMA bf16 ----------------
__global__ __launch_bounds__(32) void k_ctx(const unsigned short* __restrict__ attn,
                                            const unsigned short* __restrict__ hT,
                                            float* __restrict__ ctx) {
    int b = blockIdx.x, e0 = blockIdx.y * 16;
    int lane = threadIdx.x, r16 = lane & 15, hi = lane >> 4;
    const unsigned short* arow = attn + ((size_t)b * 16 + r16) * SPAD + hi * 8;
    const unsigned short* brow = hT + ((size_t)b * ED + e0 + r16) * SPAD + hi * 8;
    v8f c = {0.f, 0.f, 0.f, 0.f, 0.f, 0.f, 0.f, 0.f};
    for (int s0 = 0; s0 < SPAD; s0 += 32) {
        v8bf alo = *(const v8bf*)(arow + s0);
        v8bf ahi = *(const v8bf*)(arow + s0 + 16);
        v8bf blo = *(const v8bf*)(brow + s0);
        v8bf bhi = *(const v8bf*)(brow + s0 + 16);
        c = __builtin_amdgcn_wmma_f32_16x16x32_bf16(false, cat8(alo, ahi),
                                                    false, cat8(blo, bhi),
                                                    (short)0, c, false, false);
    }
#pragma unroll
    for (int r = 0; r < 8; ++r) {
        int m = r + hi * 8;                          // head row
        if (m < NH) ctx[((size_t)b * NH + m) * ED + e0 + r16] = c[r];
    }
}

// ---------------- tail of layer: Wv/Wo projection + DyT + FF + DyT (per batch) ----------------
__global__ __launch_bounds__(256) void k_mha_ff(const float* __restrict__ ctx,
                                                float* __restrict__ q,
                                                const float* __restrict__ Wv_l,
                                                const float* __restrict__ Wo_l,
                                                const float* __restrict__ d1a, const float* __restrict__ d1g,
                                                const float* __restrict__ d1b,
                                                const float* __restrict__ fW1, const float* __restrict__ fb1,
                                                const float* __restrict__ fW2, const float* __restrict__ fb2,
                                                const float* __restrict__ d2a, const float* __restrict__ d2g,
                                                const float* __restrict__ d2b) {
    int b = blockIdx.x, tid = threadIdx.x;
    __shared__ float ctxs[NH * ED];
    __shared__ float heads[NH * KDH];
    __shared__ float xs[QDIM];
    __shared__ float ffh[FFD];
    for (int i = tid; i < NH * ED; i += 256) ctxs[i] = ctx[(size_t)b * NH * ED + i];
    __syncthreads();
    for (int i = tid; i < NH * KDH; i += 256) {      // heads = ctx @ Wv
        int h = i >> 7, kd = i & (KDH - 1);
        const float* w = Wv_l + (size_t)h * ED * KDH + kd;
        float acc = 0.f;
        for (int e = 0; e < ED; ++e) acc += ctxs[h * ED + e] * w[(size_t)e * KDH];
        heads[i] = acc;
    }
    __syncthreads();
    float a1 = d1a[0];
    for (int i = tid; i < QDIM; i += 256) {          // mha_out + residual + DyT1
        float acc = 0.f;
        for (int h = 0; h < NH; ++h) {
            const float* w = Wo_l + (size_t)h * KDH * QDIM + i;
            const float* hh = heads + h * KDH;
            for (int kd = 0; kd < KDH; ++kd) acc += hh[kd] * w[(size_t)kd * QDIM];
        }
        float x = q[(size_t)b * QDIM + i] + acc;
        xs[i] = tanhf(a1 * x) * d1g[i] + d1b[i];
    }
    __syncthreads();
    for (int j = tid; j < FFD; j += 256) {           // FF up + relu
        const float* w = fW1 + j;
        float acc = fb1[j];
        for (int d = 0; d < QDIM; ++d) acc += xs[d] * w[(size_t)d * FFD];
        ffh[j] = fmaxf(acc, 0.f);
    }
    __syncthreads();
    float a2 = d2a[0];
    for (int i = tid; i < QDIM; i += 256) {          // FF down + residual + DyT2 -> new q
        const float* w = fW2 + i;
        float acc = fb2[i];
        for (int j = 0; j < FFD; ++j) acc += ffh[j] * w[(size_t)j * QDIM];
        float y = xs[i] + acc;
        q[(size_t)b * QDIM + i] = tanhf(a2 * y) * d2g[i] + d2b[i];
    }
}

// ---------------- final: w[b] = lp_Wk @ (lp_Wq^T q[b]), folded 1/sqrt(E) ----------------
__global__ __launch_bounds__(256) void k_finalw(const float* __restrict__ q,
                                                const float* __restrict__ lpWq,
                                                const float* __restrict__ lpWk,
                                                float* __restrict__ w) {
    int b = blockIdx.x, tid = threadIdx.x;
    __shared__ float qs[QDIM];
    __shared__ float us[ED];
    for (int d = tid; d < QDIM; d += 256) qs[d] = q[(size_t)b * QDIM + d];
    __syncthreads();
    for (int e = tid; e < ED; e += 256) {            // u = q @ lp_Wq
        float acc = 0.f;
        for (int d = 0; d < QDIM; ++d) acc += qs[d] * lpWq[(size_t)d * ED + e];
        us[e] = acc;
    }
    __syncthreads();
    const float sc = 0.044194173824159216f;          // 1/sqrt(E)
    for (int e = tid; e < ED; e += 256) {            // w = lp_Wk @ u
        float acc = 0.f;
        for (int ep = 0; ep < ED; ++ep) acc += lpWk[(size_t)e * ED + ep] * us[ep];
        w[(size_t)b * ED + e] = acc * sc;
    }
}

__global__ __launch_bounds__(128) void k_logits(const unsigned short* __restrict__ hmat,
                                                const float* __restrict__ w,
                                                const int* __restrict__ va,
                                                float* __restrict__ out) {
    int blk = blockIdx.x, b = blk / STOT, s = blk % STOT, tid = threadIdx.x;
    const unsigned short* hrow = hmat + ((size_t)b * SPAD + s) * ED;
    const float* wb = w + (size_t)b * ED;
    __shared__ float red[128];
    float acc = 0.f;
    for (int e = tid; e < ED; e += 128) acc += bf2f(hrow[e]) * wb[e];
    red[tid] = acc; __syncthreads();
    for (int o = 64; o > 0; o >>= 1) { if (tid < o) red[tid] += red[tid + o]; __syncthreads(); }
    if (tid == 0) {
        bool valid = (s >= NNUR) && (va[(size_t)b * (STOT - NNUR) + (s - NNUR)] != 0);
        out[(size_t)b * STOT + s] = valid ? 10.0f * tanhf(red[0]) : NEGV;
    }
}

__global__ __launch_bounds__(256) void k_lse(float* __restrict__ out) {
    int b = blockIdx.x, tid = threadIdx.x;
    float vals[9];
    int n = 0;
    float mx = -3.0e38f;
    for (int s = tid; s < STOT; s += 256) { float v = out[(size_t)b * STOT + s]; vals[n++] = v; mx = fmaxf(mx, v); }
    __shared__ float red[256];
    red[tid] = mx; __syncthreads();
    for (int o = 128; o > 0; o >>= 1) { if (tid < o) red[tid] = fmaxf(red[tid], red[tid + o]); __syncthreads(); }
    mx = red[0]; __syncthreads();
    float sm = 0.f;
    for (int i = 0; i < n; ++i) sm += __expf(vals[i] - mx);
    red[tid] = sm; __syncthreads();
    for (int o = 128; o > 0; o >>= 1) { if (tid < o) red[tid] += red[tid + o]; __syncthreads(); }
    float lse = mx + __logf(red[0]);
    n = 0;
    for (int s = tid; s < STOT; s += 256) out[(size_t)b * STOT + s] = vals[n++] - lse;
}

extern "C" void kernel_launch(void* const* d_in, const int* in_sizes, int n_in,
                              void* d_out, int out_size, void* d_ws, size_t ws_size,
                              hipStream_t stream) {
    const float* emb  = (const float*)d_in[0];
    const float* embm = (const float*)d_in[1];
    const float* noop = (const float*)d_in[2];
    const float* Wq   = (const float*)d_in[3];
    const float* Wk   = (const float*)d_in[4];
    const float* Wv   = (const float*)d_in[5];
    const float* Wo   = (const float*)d_in[6];
    const float* d1a  = (const float*)d_in[7];
    const float* d1g  = (const float*)d_in[8];
    const float* d1b  = (const float*)d_in[9];
    const float* fW1  = (const float*)d_in[10];
    const float* fb1  = (const float*)d_in[11];
    const float* fW2  = (const float*)d_in[12];
    const float* fb2  = (const float*)d_in[13];
    const float* d2a  = (const float*)d_in[14];
    const float* d2g  = (const float*)d_in[15];
    const float* d2b  = (const float*)d_in[16];
    const float* lpWq = (const float*)d_in[17];
    const float* lpWk = (const float*)d_in[18];
    const int*   va   = (const int*)d_in[19];
    const int*   nidx = (const int*)d_in[20];

    char* ws = (char*)d_ws;
    size_t off = 0;
    auto alloc = [&](size_t bytes) { size_t o = off; off = (off + bytes + 255) & ~(size_t)255; return o; };
    unsigned short* hmat = (unsigned short*)(ws + alloc((size_t)NB * SPAD * ED * 2));
    unsigned short* hT   = (unsigned short*)(ws + alloc((size_t)NB * SPAD * ED * 2));
    unsigned short* qkt  = (unsigned short*)(ws + alloc((size_t)NB * 16 * ED * 2));
    unsigned short* attn = (unsigned short*)(ws + alloc((size_t)NB * 16 * SPAD * 2));
    float* scores = (float*)(ws + alloc((size_t)NB * NH * SPAD * 4));
    float* ctx    = (float*)(ws + alloc((size_t)NB * NH * ED * 4));
    float* q      = (float*)(ws + alloc((size_t)NB * QDIM * 4));
    float* wvec   = (float*)(ws + alloc((size_t)NB * ED * 4));
    float* out    = (float*)d_out;

    k_prep_h<<<NB * SPAD, 256, 0, stream>>>(emb, noop, hmat);
    k_prep_q<<<NB, 256, 0, stream>>>(emb, embm, noop, nidx, q);
    k_transpose<<<dim3(NB, SPAD / 16, ED / 16), 256, 0, stream>>>(hmat, hT);

    for (int l = 0; l < NL; ++l) {
        const float* Wq_l = Wq + (size_t)l * NH * QDIM * KDH;
        const float* Wk_l = Wk + (size_t)l * NH * ED * KDH;
        const float* Wv_l = Wv + (size_t)l * NH * ED * KDH;
        const float* Wo_l = Wo + (size_t)l * NH * KDH * QDIM;
        k_qk<<<dim3(NB, 16), 128, 0, stream>>>(q, Wq_l, Wk_l, qkt);
        k_scores<<<dim3(NB, SPAD / 16), 32, 0, stream>>>(hmat, qkt, va, scores);
        k_softmax<<<dim3(NB, 16), 256, 0, stream>>>(scores, attn);
        k_ctx<<<dim3(NB, ED / 16), 32, 0, stream>>>(attn, hT, ctx);
        k_mha_ff<<<NB, 256, 0, stream>>>(ctx, q, Wv_l, Wo_l,
                                         d1a + l, d1g + (size_t)l * QDIM, d1b + (size_t)l * QDIM,
                                         fW1 + (size_t)l * QDIM * FFD, fb1 + (size_t)l * FFD,
                                         fW2 + (size_t)l * FFD * QDIM, fb2 + (size_t)l * QDIM,
                                         d2a + l, d2g + (size_t)l * QDIM, d2b + (size_t)l * QDIM);
    }

    k_finalw<<<NB, 256, 0, stream>>>(q, lpWq, lpWk, wvec);
    k_logits<<<NB * STOT, 128, 0, stream>>>(hmat, wvec, va, out);
    k_lse<<<NB, 256, 0, stream>>>(out);
    (void)in_sizes; (void)n_in; (void)out_size; (void)ws_size;
}
